// GlobalAttentionPool_26946624815832
// MI455X (gfx1250) — matrix operand order repaired and checked by reading, not compiled, not run on hardware
//
#include <hip/hip_runtime.h>

typedef unsigned short u16;
typedef unsigned int   u32;

typedef __attribute__((ext_vector_type(16))) __bf16 v16bf;
typedef __attribute__((ext_vector_type(8)))  float  v8f;

#define Hdim 256
#define H2   128
#define KT   8     // 256 / 32
#define NT   8     // 128 / 16
#define XPAD 34    // padded bf16 row length for the X tile in LDS
#define POOL_ROWS 512

union FragU {
    v16bf  v;
    u32    u[8];
    float4 f4[2];
};

__device__ __forceinline__ u16 f2bf(float f) {
    u32 u = __float_as_uint(f);
    u32 r = u + 0x7FFFu + ((u >> 16) & 1u);   // round-to-nearest-even
    return (u16)(r >> 16);
}

__device__ __forceinline__ u32 mapf(float f) {   // monotonic float->uint
    u32 u = __float_as_uint(f);
    return (u & 0x80000000u) ? ~u : (u | 0x80000000u);
}
__device__ __forceinline__ float unmapf(u32 m) {
    u32 u = (m & 0x80000000u) ? (m ^ 0x80000000u) : ~m;
    return __uint_as_float(u);
}

// ---------------------------------------------------------------- init ------
__global__ void init_kernel(float* out, int outn, u32* umax, float* denom, int B) {
    int i = blockIdx.x * blockDim.x + threadIdx.x;
    if (i < outn) out[i] = 0.0f;
    if (i < B) { umax[i] = 0u; denom[i] = 0.0f; }   // 0 < mapf(any finite float)
}

// -------------------------------------------- repack W1 into B-fragments ----
// w1frag[kt][nt][lane][e] : lane 0-15 -> col N = nt*16 + lane, K = kt*32 + e
//                           lane16-31 -> col N = nt*16 + lane-16, K = kt*32 + 16 + e
__global__ void prep_w1(const float* __restrict__ W1, u16* __restrict__ w1frag) {
    int idx = blockIdx.x * blockDim.x + threadIdx.x;   // 0 .. 8*8*32-1
    if (idx >= KT * NT * 32) return;
    int lane = idx & 31;
    int nt   = (idx >> 5) & 7;
    int kt   = idx >> 8;
    int n     = nt * 16 + (lane & 15);
    int kbase = kt * 32 + (lane >> 4) * 16;
    u16* dst = w1frag + idx * 16;
#pragma unroll
    for (int e = 0; e < 16; ++e)
        dst[e] = f2bf(W1[(size_t)(kbase + e) * H2 + n]);
}

// -------------------------------------------------- fused score GEMM --------
// scores[i] = tanh(x[i]@W1 + b1) @ W2 + b2, bf16 WMMA, fp32 accumulate.
__global__ __launch_bounds__(256) void score_kernel(
        const float* __restrict__ x, const u16* __restrict__ w1frag,
        const float* __restrict__ b1, const float* __restrict__ W2,
        const float* __restrict__ b2v, float* __restrict__ scores, int nrows) {
    __shared__ u16 xs[128 * XPAD];   // 8.5 KB bf16 X tile

    const int tid  = threadIdx.x;
    const int wave = tid >> 5;
    const int lane = tid & 31;
    const int half = lane >> 4;
    const int lm   = lane & 15;
    const long rowBase = (long)blockIdx.x * 128;

    v8f acc[NT] = {};
    const float b2s = b2v[0];

    for (int kt = 0; kt < KT; ++kt) {
        // Issue ALL B-fragment loads for this K-step first: their L2 latency
        // is hidden under the X-tile staging + barrier below, and the 8 WMMAs
        // can then issue back-to-back after a single wait.
        FragU b[NT];
#pragma unroll
        for (int nt = 0; nt < NT; ++nt) {
            const float4* bp =
                (const float4*)(w1frag + ((size_t)(kt * NT + nt) * 32 + lane) * 16);
            b[nt].f4[0] = bp[0];
            b[nt].f4[1] = bp[1];
        }

        __syncthreads();   // xs safe to overwrite
        {   // stage 128x32 fp32 -> bf16 LDS, fully coalesced (8 thr * float4 per row)
            int tr = tid >> 3;
            int tc = (tid & 7) * 4;
#pragma unroll
            for (int it = 0; it < 4; ++it) {
                int  r    = tr + it * 32;
                long grow = rowBase + r;
                float4 v = make_float4(0.f, 0.f, 0.f, 0.f);
                if (grow < nrows)
                    v = *(const float4*)(x + grow * Hdim + kt * 32 + tc);
                u16* dst = &xs[r * XPAD + tc];
                dst[0] = f2bf(v.x); dst[1] = f2bf(v.y);
                dst[2] = f2bf(v.z); dst[3] = f2bf(v.w);
            }
        }
        __syncthreads();

        // A fragment: lane 0-15 row M=lm K=0..15; lane 16-31 K=16..31
        FragU a;
        int arow = wave * 16 + lm;
#pragma unroll
        for (int d = 0; d < 8; ++d) {
            int ke = 2 * d + 16 * half;
            a.u[d] = *(const u32*)&xs[arow * XPAD + ke];
        }

#pragma unroll
        for (int nt = 0; nt < NT; ++nt)
            acc[nt] = __builtin_amdgcn_wmma_f32_16x16x32_bf16(
                false, a.v, false, b[nt].v, (short)0, acc[nt], false, false);
    }

    // Epilogue: score = sum_n tanh(acc + b1[n]) * W2[n] ; acc VGPR r holds
    // (row = r + 8*half, col = lm) of each 16x16 tile.
    float part[8];
#pragma unroll
    for (int r = 0; r < 8; ++r) part[r] = 0.0f;
#pragma unroll
    for (int nt = 0; nt < NT; ++nt) {
        int   col = nt * 16 + lm;
        float w2c = W2[col];
        float bc  = b1[col];
#pragma unroll
        for (int r = 0; r < 8; ++r)
            part[r] += tanhf(acc[nt][r] + bc) * w2c;
    }
#pragma unroll
    for (int m = 1; m < 16; m <<= 1)
#pragma unroll
        for (int r = 0; r < 8; ++r)
            part[r] += __shfl_xor(part[r], m, 32);

    if (lm == 0) {
#pragma unroll
        for (int r = 0; r < 8; ++r) {
            long row = rowBase + wave * 16 + half * 8 + r;
            if (row < nrows) scores[row] = part[r] + b2s;
        }
    }
}

// ------------------------------------------------ segment max (atomic) ------
__global__ void segmax_kernel(const float* __restrict__ scores,
                              const int* __restrict__ batch,
                              u32* __restrict__ umax, int nrows) {
    int i = blockIdx.x * blockDim.x + threadIdx.x;
    if (i >= nrows) return;
    atomicMax(&umax[batch[i]], mapf(scores[i]));
}

// ------------------------------------------- exp + segment sum (atomic) -----
__global__ void expsum_kernel(float* __restrict__ score_e,
                              const int* __restrict__ batch,
                              const u32* __restrict__ umax,
                              float* __restrict__ denom, int nrows) {
    int i = blockIdx.x * blockDim.x + threadIdx.x;
    if (i >= nrows) return;
    int   b = batch[i];
    float e = __expf(score_e[i] - unmapf(umax[b]));
    score_e[i] = e;                 // overwrite score with exp in place
    atomicAdd(&denom[b], e);
}

// ----------------------------- weighted segment pooling (sorted batch) ------
__global__ __launch_bounds__(256) void pool_kernel(
        const float* __restrict__ x, const int* __restrict__ batch,
        const float* __restrict__ e, const float* __restrict__ denom,
        float* __restrict__ out, int nrows) {
    const int h  = threadIdx.x;                 // one thread per feature
    long r0 = (long)blockIdx.x * POOL_ROWS;
    long r1 = r0 + POOL_ROWS;
    if (r1 > nrows) r1 = nrows;
    if (r0 >= r1) return;

    float acc   = 0.0f;
    int   prevb = batch[r0];
    for (long r = r0; r < r1; ++r) {
        if (r + 4 < r1)
            __builtin_prefetch(x + (r + 4) * Hdim + h, 0, 1);
        int   b = batch[r];
        float p = e[r] / denom[b];
        if (b != prevb) {                       // segment boundary: flush
            atomicAdd(&out[(size_t)prevb * Hdim + h], acc);
            acc   = 0.0f;
            prevb = b;
        }
        acc += x[r * Hdim + h] * p;
    }
    atomicAdd(&out[(size_t)prevb * Hdim + h], acc);
}

// ---------------------------------------------------------------------------
extern "C" void kernel_launch(void* const* d_in, const int* in_sizes, int n_in,
                              void* d_out, int out_size, void* d_ws, size_t ws_size,
                              hipStream_t stream) {
    const float* x     = (const float*)d_in[0];
    const int*   batch = (const int*)  d_in[1];
    const float* W1    = (const float*)d_in[2];
    const float* b1    = (const float*)d_in[3];
    const float* W2    = (const float*)d_in[4];
    const float* b2    = (const float*)d_in[5];
    float*       out   = (float*)d_out;

    const int nrows = in_sizes[1];        // N (batch index array length)
    const int B     = out_size / Hdim;    // number of segments

    // workspace layout (all 256B-aligned offsets)
    char*  ws      = (char*)d_ws;
    u16*   w1frag  = (u16*)  ws;                     // 64 KB
    u32*   umax    = (u32*)  (ws + 65536);           // B u32
    float* denom   = (float*)(ws + 65536 + 4096);    // B floats
    float* score_e = (float*)(ws + 65536 + 8192);    // N floats (scores, then exp)

    int initN = out_size > B ? out_size : B;
    init_kernel<<<(initN + 255) / 256, 256, 0, stream>>>(out, out_size, umax, denom, B);
    prep_w1<<<(KT * NT * 32 + 255) / 256, 256, 0, stream>>>(W1, w1frag);

    int gemmBlocks = (nrows + 127) / 128;
    score_kernel<<<gemmBlocks, 256, 0, stream>>>(x, w1frag, b1, W2, b2, score_e, nrows);

    segmax_kernel<<<(nrows + 255) / 256, 256, 0, stream>>>(score_e, batch, umax, nrows);
    expsum_kernel<<<(nrows + 255) / 256, 256, 0, stream>>>(score_e, batch, umax, denom, nrows);

    int poolBlocks = (nrows + POOL_ROWS - 1) / POOL_ROWS;
    pool_kernel<<<poolBlocks, 256, 0, stream>>>(x, batch, score_e, denom, out, nrows);
}
